// LSTMModel_51668456571013
// MI455X (gfx1250) — compile-verified, matrix operand order checked by reading, not run on hardware
//
#include <hip/hip_runtime.h>
#include <hip/hip_bf16.h>

// ---------------------------------------------------------------------------
// LSTM (T=256,B=256,E=256,H=512) + dense projection to V=50257, CDNA5 WMMA.
// f16 multiplicands, f32 WMMA accumulation, f32 cell state & output.
// Staging uses gfx1250 async global->LDS DMA (ASYNCcnt) + double buffering.
// ---------------------------------------------------------------------------

#define T_ 256
#define B_ 256
#define E_ 256
#define H_ 512
#define V_ 50257
#define KTOT_ 768     // E + H
#define FOURH_ 2048   // 4*H
#define LDA_ 40       // LDS row stride in halves (80B: 16B-aligned, bank-spread)
#define BUFB_ 10240   // bytes per LDS double-buffer slot (As 5120 + Bs 5120)

typedef _Float16 v8h  __attribute__((ext_vector_type(8)));
typedef _Float16 v16h __attribute__((ext_vector_type(16)));
typedef float    v8f  __attribute__((ext_vector_type(8)));

__device__ __forceinline__ float sigf(float x) {
    return 1.0f / (1.0f + __expf(-x));
}

// Async DMA: copy 16 bytes global -> LDS per lane (GLOBAL_LOAD_ASYNC_TO_LDS_B128,
// tracked by ASYNCcnt; bypasses VGPRs). lds_addr = wave-relative LDS byte offset
// (= low 32 bits of the generic shared-space pointer per CDNA5 aperture rules).
__device__ __forceinline__ void async_copy16(unsigned lds_addr, const void* gptr) {
    asm volatile("global_load_async_to_lds_b128 %0, %1, off"
                 :
                 : "v"(lds_addr), "v"((unsigned long long)(uintptr_t)gptr)
                 : "memory");
}
__device__ __forceinline__ void wait_async0() {
    asm volatile("s_wait_asynccnt 0x0" ::: "memory");
}

// Load a 16-half WMMA A/B fragment for this lane from an LDS row.
// Per CDNA5 ISA 16-bit A layout: lane-group g=lane/16 owns K-halves
// [8g, 8g+8) and [16+8g, 24+8g) -> two contiguous 16B runs.
__device__ __forceinline__ v16h load_frag(const _Float16* row, int grp) {
    v8h a = *(const v8h*)(row + grp * 8);
    v8h b = *(const v8h*)(row + 16 + grp * 8);
    v16h r;
#pragma unroll
    for (int i = 0; i < 8; ++i) { r[i] = a[i]; r[i + 8] = b[i]; }
    return r;
}

// ---------------------------------------------------------------------------
// One-time: transpose-convert lstm_kernel [768,2048] f32 -> kT [2048][768] f16
// ---------------------------------------------------------------------------
__global__ __launch_bounds__(256) void k_convert(const float* __restrict__ w,
                                                 _Float16* __restrict__ kT) {
    int i = blockIdx.x * 256 + threadIdx.x;        // i over 768*2048
    if (i < KTOT_ * FOURH_) {
        int kk = i / FOURH_;                       // row (K)     coalesced read
        int cc = i - kk * FOURH_;                  // col (4H)
        kT[(size_t)cc * KTOT_ + kk] = (_Float16)w[i];
    }
}

// One-time: convert x [T,B,E] f32 -> f16 (8 elements / thread, exact grid)
__global__ __launch_bounds__(256) void k_convert_x(const float* __restrict__ x,
                                                   _Float16* __restrict__ xh) {
    size_t i = ((size_t)blockIdx.x * 256 + threadIdx.x) * 8;
    float4 f0 = *(const float4*)(x + i);
    float4 f1 = *(const float4*)(x + i + 4);
    v8h o;
    o[0] = (_Float16)f0.x; o[1] = (_Float16)f0.y;
    o[2] = (_Float16)f0.z; o[3] = (_Float16)f0.w;
    o[4] = (_Float16)f1.x; o[5] = (_Float16)f1.y;
    o[6] = (_Float16)f1.z; o[7] = (_Float16)f1.w;
    *(v8h*)(xh + i) = o;
}

__global__ __launch_bounds__(256) void k_init(_Float16* __restrict__ h0,
                                              float* __restrict__ c) {
    int i = blockIdx.x * 256 + threadIdx.x;        // over B*H
    if (i < B_ * H_) { h0[i] = (_Float16)0.0f; c[i] = 0.0f; }
}

// ---------------------------------------------------------------------------
// One LSTM timestep, fused: gates GEMM ([B,768]x[768,2048]) + cell update.
// Block = 256 threads (8 waves). Block tile: 64 batch rows x 16 H-columns,
// covering the matching 16-col slice of all 4 gates (i,j,f,o).
// Waves 0-3: gates {i,j} for M-subtile w; waves 4-7: gates {f,o}.
// K-loop: double-buffered LDS, async-DMA staging overlapped with WMMA.
// ---------------------------------------------------------------------------
__global__ __launch_bounds__(256) void k_step(const _Float16* __restrict__ xh,
                                              const _Float16* __restrict__ kT,
                                              const float* __restrict__ bias,
                                              const _Float16* __restrict__ hin,
                                              _Float16* __restrict__ hout,
                                              float* __restrict__ c,
                                              int t) {
    __shared__ __align__(16) unsigned char smem[2 * BUFB_];   // 20 KB
    float* gacc = (float*)smem;                    // [4][64][16] f32 (aliased)

    const int tid  = threadIdx.x;
    const int lane = tid & 31;
    const int wave = tid >> 5;
    const int wm   = wave & 3;                      // M subtile (16 rows)
    const int gp   = wave >> 2;                     // gate pair: 0->{i,j} 1->{f,o}
    const int lrow = lane & 15;
    const int grp  = lane >> 4;

    const int m0 = blockIdx.x * 64;                 // batch tile base
    const int n0 = blockIdx.y * 16;                 // H-column tile base

    const int r  = tid & 63;                        // staging row 0..63
    const int k8 = (tid >> 6) << 3;                 // staging K sub-run 0/8/16/24

    // staging row r of Bs holds weight column: gate (r/16), col n0+(r%16)
    const int colg = (r >> 4) * H_ + n0 + (r & 15);
    const _Float16* bsrc   = kT + (size_t)colg * KTOT_;
    const _Float16* arow_x = xh + ((size_t)t * B_ + (m0 + r)) * E_;
    const _Float16* arow_h = hin + (size_t)(m0 + r) * H_;

    const unsigned ldsbase = (unsigned)(uintptr_t)smem;
    const unsigned stgoff  = (unsigned)(r * (LDA_ * 2) + k8 * 2); // byte offset in slot

    // issue this thread's two 16B async copies for K-chunk kc into buffer slot b
    auto stage = [&](int kc, int b) {
        const int k0 = kc * 32;
        const _Float16* asrc =
            (k0 < E_) ? (arow_x + k0 + k8) : (arow_h + (k0 - E_) + k8);
        const unsigned slot = ldsbase + (unsigned)(b * BUFB_) + stgoff;
        async_copy16(slot, asrc);                   // As half of slot
        async_copy16(slot + 5120u, bsrc + k0 + k8); // Bs half of slot
    };

    v8f acc0 = {};
    v8f acc1 = {};

    stage(0, 0);
    wait_async0();
    __syncthreads();

    for (int kc = 0; kc < KTOT_ / 32; ++kc) {
        const int cur = kc & 1;
        if (kc + 1 < KTOT_ / 32) stage(kc + 1, cur ^ 1);  // prefetch next chunk

        _Float16* AsC = (_Float16*)(smem + cur * BUFB_);
        _Float16* BsC = AsC + 64 * LDA_;
        v16h a  = load_frag(AsC + (16 * wm + lrow) * LDA_, grp);
        v16h b0 = load_frag(BsC + ((gp * 2 + 0) * 16 + lrow) * LDA_, grp);
        v16h b1 = load_frag(BsC + ((gp * 2 + 1) * 16 + lrow) * LDA_, grp);
        acc0 = __builtin_amdgcn_wmma_f32_16x16x32_f16(false, a, false, b0,
                                                      (short)0, acc0, false, false);
        acc1 = __builtin_amdgcn_wmma_f32_16x16x32_f16(false, a, false, b1,
                                                      (short)0, acc1, false, false);

        wait_async0();       // next chunk's DMA has landed
        __syncthreads();     // whole WG done reading cur / sees next buffer
    }

    // ---- exchange gate tiles through LDS, then fused cell update ----
#pragma unroll
    for (int rr = 0; rr < 8; ++rr) {
        const int m = 16 * wm + rr + 8 * grp;       // C/D layout: M = vgpr + 8*(lane/16)
        gacc[(gp * 2 + 0) * 1024 + m * 16 + lrow] = acc0[rr];
        gacc[(gp * 2 + 1) * 1024 + m * 16 + lrow] = acc1[rr];
    }
    __syncthreads();

#pragma unroll
    for (int it = 0; it < 4; ++it) {
        const int idx = tid + it * 256;             // 1024 (m,n) cells / block
        const int m = idx >> 4;
        const int n = idx & 15;
        const int b  = m0 + m;
        const int hc = n0 + n;
        const float gi = gacc[0 * 1024 + m * 16 + n] + bias[0 * H_ + hc];
        const float gj = gacc[1 * 1024 + m * 16 + n] + bias[1 * H_ + hc];
        const float gf = gacc[2 * 1024 + m * 16 + n] + bias[2 * H_ + hc];
        const float go = gacc[3 * 1024 + m * 16 + n] + bias[3 * H_ + hc];
        const size_t ci = (size_t)b * H_ + hc;
        const float cold = c[ci];
        const float cn = sigf(gf + 1.0f) * cold + sigf(gi) * tanhf(gj);
        const float hn = sigf(go) * tanhf(cn);
        c[ci] = cn;
        hout[ci] = (_Float16)hn;
    }
}

// ---------------------------------------------------------------------------
// Projection: out[256,50257] = h_last(f16) @ W_out(f32->f16) + b_out, WMMA.
// Block tile 64 x 64 (waves: 4 M-subtiles x 2 V-subtiles each).
// A staged via async DMA; B converted f32->f16 in flight.
// ---------------------------------------------------------------------------
__global__ __launch_bounds__(256) void k_proj(const _Float16* __restrict__ hlast,
                                              const float* __restrict__ Wout,
                                              const float* __restrict__ bout,
                                              float* __restrict__ out) {
    __shared__ __align__(16) unsigned char smem[BUFB_];
    _Float16* As = (_Float16*)smem;                 // [64][LDA_]
    _Float16* Bs = As + 64 * LDA_;                  // [64][LDA_]

    const int tid  = threadIdx.x;
    const int lane = tid & 31;
    const int wave = tid >> 5;
    const int wm   = wave & 3;
    const int vp   = wave >> 2;                     // V-subtile pair base = vp*2
    const int lrow = lane & 15;
    const int grp  = lane >> 4;

    const int m0 = blockIdx.x * 64;
    const int v0 = blockIdx.y * 64;

    const int r  = tid & 63;
    const int k8 = (tid >> 6) << 3;
    const int vcolr = v0 + r;                       // staging column of W_out

    const unsigned ldsA = (unsigned)(uintptr_t)smem + (unsigned)(r * (LDA_ * 2) + k8 * 2);

    v8f acc0 = {};
    v8f acc1 = {};

    for (int kc = 0; kc < H_ / 32; ++kc) {
        const int k0 = kc * 32;
        __syncthreads();
        // stage A: h_last rows, async DMA (f16 already)
        async_copy16(ldsA, hlast + (size_t)(m0 + r) * H_ + k0 + k8);
        // stage B: W_out column vcolr, K-run k0+k8..+8 (coalesced across lanes)
        _Float16* db = Bs + r * LDA_ + k8;
        if (vcolr < V_) {
#pragma unroll
            for (int j = 0; j < 8; ++j)
                db[j] = (_Float16)Wout[(size_t)(k0 + k8 + j) * V_ + vcolr];
        } else {
#pragma unroll
            for (int j = 0; j < 8; ++j) db[j] = (_Float16)0.0f;
        }
        wait_async0();
        __syncthreads();
        v16h a  = load_frag(As + (16 * wm + lrow) * LDA_, grp);
        v16h b0 = load_frag(Bs + ((vp * 2 + 0) * 16 + lrow) * LDA_, grp);
        v16h b1 = load_frag(Bs + ((vp * 2 + 1) * 16 + lrow) * LDA_, grp);
        acc0 = __builtin_amdgcn_wmma_f32_16x16x32_f16(false, a, false, b0,
                                                      (short)0, acc0, false, false);
        acc1 = __builtin_amdgcn_wmma_f32_16x16x32_f16(false, a, false, b1,
                                                      (short)0, acc1, false, false);
    }

#pragma unroll
    for (int rr = 0; rr < 8; ++rr) {
        const int m = m0 + 16 * wm + rr + 8 * grp;
        const int vc0 = v0 + (vp * 2 + 0) * 16 + lrow;
        const int vc1 = v0 + (vp * 2 + 1) * 16 + lrow;
        if (vc0 < V_) out[(size_t)m * V_ + vc0] = acc0[rr] + bout[vc0];
        if (vc1 < V_) out[(size_t)m * V_ + vc1] = acc1[rr] + bout[vc1];
    }
}

// ---------------------------------------------------------------------------
extern "C" void kernel_launch(void* const* d_in, const int* in_sizes, int n_in,
                              void* d_out, int out_size, void* d_ws, size_t ws_size,
                              hipStream_t stream) {
    (void)in_sizes; (void)n_in; (void)out_size; (void)ws_size;
    const float* x         = (const float*)d_in[0];   // [T,B,E]
    const float* lstm_k    = (const float*)d_in[1];   // [768, 2048]
    const float* lstm_bias = (const float*)d_in[2];   // [2048]
    const float* W_out     = (const float*)d_in[3];   // [H, V]
    const float* b_out     = (const float*)d_in[4];   // [V]
    float* out             = (float*)d_out;           // [B, V]

    // workspace layout (~38 MB total)
    char* ws = (char*)d_ws;
    _Float16* kT    = (_Float16*)ws;                        // 2048*768*2  =  3,145,728
    _Float16* xh    = (_Float16*)(ws + 3145728);            // T*B*E*2     = 33,554,432
    _Float16* hbuf0 = (_Float16*)(ws + 36700160);           // 256*512*2   =    262,144
    _Float16* hbuf1 = (_Float16*)(ws + 36962304);           //                  262,144
    float*    cbuf  = (float*)   (ws + 37224448);           // 256*512*4   =    524,288

    // 1) one-time weight transpose/convert, x convert, state init
    k_convert<<<(KTOT_ * FOURH_ + 255) / 256, 256, 0, stream>>>(lstm_k, kT);
    k_convert_x<<<(T_ * B_ * E_) / (256 * 8), 256, 0, stream>>>(x, xh);
    k_init<<<(B_ * H_ + 255) / 256, 256, 0, stream>>>(hbuf0, cbuf);

    // 2) serial recurrence: one fused kernel per timestep (ping-pong h)
    for (int t = 0; t < T_; ++t) {
        const _Float16* hin = (t & 1) ? hbuf1 : hbuf0;
        _Float16* hout      = (t & 1) ? hbuf0 : hbuf1;
        k_step<<<dim3(B_ / 64, H_ / 16), 256, 0, stream>>>(xh, kT, lstm_bias,
                                                           hin, hout, cbuf, t);
    }
    // after 256 steps the last write landed in hbuf0

    // 3) projection to vocab
    k_proj<<<dim3(B_ / 64, (V_ + 63) / 64), 256, 0, stream>>>(hbuf0, W_out,
                                                              b_out, out);
}
